// Critic_QuadAdv_MultiheadAttention_80169859547823
// MI455X (gfx1250) — matrix operand order repaired and checked by reading, not compiled, not run on hardware
//
#include <hip/hip_runtime.h>
#include <hip/hip_bf16.h>

typedef __attribute__((ext_vector_type(16))) _Float16 v16h;
typedef __attribute__((ext_vector_type(8)))  _Float16 v8h;
typedef __attribute__((ext_vector_type(8)))  float    v8f;

constexpr int BATCH      = 65536;
constexpr int HID        = 128;
constexpr int OBS_STRIDE = 114;   // SELF_OBS_DIM + NUM_ADV*NBR_OBS_DIM
constexpr int SELF_DIM   = 18;
constexpr int NBR_DIM    = 12;
constexpr int NADV       = 8;
constexpr int NGROUPS    = 8192;  // BATCH / NUM_AGENTS
constexpr int OUT1_ELEMS = BATCH * HID;   // multi_head_attention elems

union HF16x16 { v16h v; v8h h[2]; };

// tanh: prefer the native CDNA5 V_TANH_F32 transcendental (single trans op);
// fall back to branch-free 1 - 2/(e^{2x}+1) (v_exp_f32 + v_rcp_f32 based).
__device__ __forceinline__ float fast_tanh(float x) {
#if __has_builtin(__builtin_amdgcn_tanhf)
  return __builtin_amdgcn_tanhf(x);
#else
  float e = __builtin_amdgcn_exp2f(x * 2.8853900817779268f); // 2*log2(e)
  float r = __builtin_amdgcn_rcpf(e + 1.0f);
  return 1.0f - 2.0f * r;
#endif
}

__device__ __forceinline__ float fast_exp(float x) {
  return __builtin_amdgcn_exp2f(x * 1.4426950408889634f);    // log2(e)
}

// A-fragment (16x32 f16). LDS layout: row-major, `stride` halves per row.
// lane L holds M = L&15; halves 0..7 -> K = (L>>4)*8 + 0..7 ; halves 8..15 -> +16
__device__ __forceinline__ v16h ldA(const _Float16* base, int stride, int lane) {
  const _Float16* p = base + (lane & 15) * stride + ((lane >> 4) << 3);
  HF16x16 f;
  f.h[0] = *(const v8h*)(p);
  f.h[1] = *(const v8h*)(p + 16);
  return f.v;
}

// B-fragment (32x16 f16) from N-major (transposed) weights, `stride` halves per N-row.
// lane L holds N = L&15; halves 0..7 -> K = (L>>4)*8 + 0..7 ; halves 8..15 -> +16
__device__ __forceinline__ v16h ldB(const _Float16* baseT, int stride, int lane) {
  const _Float16* p = baseT + (lane & 15) * stride + ((lane >> 4) << 3);
  HF16x16 f;
  f.h[0] = *(const v8h*)(p);
  f.h[1] = *(const v8h*)(p + 16);
  return f.v;
}

// Load 128x128 f32 weight W[k][n] -> LDS transposed Wt[n][k] as f16.
__device__ __forceinline__ void load_wt128(_Float16* Wt, const float* __restrict__ W, int tid) {
#pragma unroll
  for (int j = 0; j < 64; ++j) {
    int idx = j * 256 + tid;          // 16384 elements
    int n = idx >> 7, k = idx & 127;
    Wt[n * 128 + k] = (_Float16)W[k * 128 + n];
  }
}

// One 128x128x128 GEMM (per block): D = tanh(A @ W + b), A rows 0..127 in LDS,
// WtT is N-major transposed weights in LDS. Each wave owns a 16-row strip.
__device__ __forceinline__ void gemm128_tanh(const _Float16* A, const _Float16* WtT,
                                             const float* __restrict__ bias,
                                             _Float16* Out, int tid) {
  const int lane = tid & 31;
  const int row0 = (tid >> 5) * 16;
  v8f acc[8];
#pragma unroll
  for (int t = 0; t < 8; ++t) acc[t] = (v8f){};
#pragma unroll
  for (int kb = 0; kb < 128; kb += 32) {
    v16h a = ldA(A + row0 * 128 + kb, 128, lane);
#pragma unroll
    for (int t = 0; t < 8; ++t) {
      v16h bm = ldB(WtT + (t * 16) * 128 + kb, 128, lane);
      acc[t] = __builtin_amdgcn_wmma_f32_16x16x32_f16(false, a, false, bm,
                                                      (short)0, acc[t], false, false);
    }
  }
  const int nlo = lane & 15;
  const int mg  = (lane >> 4) << 3;
#pragma unroll
  for (int t = 0; t < 8; ++t) {
    int n = t * 16 + nlo;
    float bv = bias[n];
#pragma unroll
    for (int v = 0; v < 8; ++v) {
      float x = fast_tanh(acc[t][v] + bv);
      Out[(row0 + mg + v) * 128 + n] = (_Float16)x;
    }
  }
}

// ---------------------------------------------------------------------------
// Kernel 1: fused embed MLP + value MLP + neighbor attention.
// Block = 256 threads (8 waves), handles 16 batches = 128 neighbor rows.
// LDS (dynamic, halves): Wt[16384] | Xs[4096] | Hs[16384] | Es[16384]  = 104KB
// ---------------------------------------------------------------------------
__global__ __launch_bounds__(256)
void fused_embed_value_attn(const float* __restrict__ obs,
                            const float* __restrict__ eW1, const float* __restrict__ eb1,
                            const float* __restrict__ eW2, const float* __restrict__ eb2,
                            const float* __restrict__ vW1, const float* __restrict__ vb1,
                            const float* __restrict__ vW2, const float* __restrict__ vb2,
                            float* __restrict__ attn_f32,      // d_out + OUT1_ELEMS
                            _Float16* __restrict__ attn_f16) { // d_ws
  extern __shared__ _Float16 smem[];
  _Float16* Wt = smem;              // 16384 halves
  _Float16* Xs = smem + 16384;      //  4096 halves
  _Float16* Hs = Xs + 4096;         // 16384 halves (hidden; later aliased as values)
  _Float16* Es = Hs + 16384;        // 16384 halves (embeddings)
  __shared__ float sc[16][8];

  const int tid  = threadIdx.x;
  const int lane = tid & 31;
  const int bid  = blockIdx.x;
  const int b0   = bid * 16;        // first batch handled by this block
  const long i0  = (long)b0 * NADV; // first neighbor-embedding row

  // ---- stage input X (128 rows x 32 K, f16): [self_obs[(i mod B)], nbr_obs, pad] ----
  {
    int r = tid >> 1;
    int cbase = (tid & 1) * 16;
    long i = i0 + r;
    int selfRow  = (int)(i & (BATCH - 1));     // the reference's tile() cross-batch pairing
    int nbrBatch = b0 + (r >> 3);
    int n        = r & 7;
    const float* selfp = obs + (long)selfRow * OBS_STRIDE;
    const float* nbrp  = obs + (long)nbrBatch * OBS_STRIDE + SELF_DIM + n * NBR_DIM;
#pragma unroll
    for (int c = 0; c < 16; ++c) {
      int col = cbase + c;
      float v = 0.f;
      if (col < SELF_DIM)                 v = selfp[col];
      else if (col < SELF_DIM + NBR_DIM)  v = nbrp[col - SELF_DIM];
      Xs[r * 32 + col] = (_Float16)v;
    }
  }
  // ---- Wt <- eW1^T, K padded 30 -> 32 ----
#pragma unroll
  for (int j = 0; j < 16; ++j) {
    int idx = j * 256 + tid;          // 4096
    int n = idx >> 5, k = idx & 31;
    float v = (k < SELF_DIM + NBR_DIM) ? eW1[k * HID + n] : 0.f;
    Wt[n * 32 + k] = (_Float16)v;
  }
  __syncthreads();

  // ---- GEMM1: h = tanh(X @ eW1 + eb1)  (K = 32, one WMMA per tile) ----
  {
    const int row0 = (tid >> 5) * 16;
    v16h a = ldA(Xs + row0 * 32, 32, lane);
    const int nlo = lane & 15;
    const int mg  = (lane >> 4) << 3;
#pragma unroll
    for (int t = 0; t < 8; ++t) {
      v16h bm = ldB(Wt + (t * 16) * 32, 32, lane);
      v8f acc = (v8f){};
      acc = __builtin_amdgcn_wmma_f32_16x16x32_f16(false, a, false, bm,
                                                   (short)0, acc, false, false);
      int n = t * 16 + nlo;
      float bv = eb1[n];
#pragma unroll
      for (int v = 0; v < 8; ++v) {
        float x = fast_tanh(acc[v] + bv);
        Hs[(row0 + mg + v) * 128 + n] = (_Float16)x;
      }
    }
  }
  __syncthreads();

  // ---- GEMM2: emb = tanh(h @ eW2 + eb2) ----
  load_wt128(Wt, eW2, tid);
  __syncthreads();
  gemm128_tanh(Hs, Wt, eb2, Es, tid);
  __syncthreads();

  // ---- GEMM3: hv = tanh(emb @ vW1 + vb1) ----
  load_wt128(Wt, vW1, tid);
  __syncthreads();
  gemm128_tanh(Es, Wt, vb1, Hs, tid);
  __syncthreads();

  // ---- GEMM4: val = tanh(hv @ vW2 + vb2)  (in-place per-strip: Vs aliases Hs) ----
  load_wt128(Wt, vW2, tid);
  __syncthreads();
  gemm128_tanh(Hs, Wt, vb2, Hs, tid);
  __syncthreads();

  // ---- attention over 8 neighbors per batch ----
  float* qf = (float*)Xs;           // reuse Xs region: 16 x 128 f32 = 8KB
#pragma unroll
  for (int j = 0; j < 8; ++j) {
    int idx = j * 256 + tid;        // 2048
    int lb = idx >> 7, c = idx & 127;
    float s = 0.f;
#pragma unroll
    for (int n = 0; n < 8; ++n) s += (float)Es[(lb * 8 + n) * 128 + c];
    qf[idx] = s * 0.125f;
  }
  __syncthreads();
  if (tid < 128) {
    int lb = tid >> 3, n = tid & 7;
    const _Float16* er = Es + (lb * 8 + n) * 128;
    const float*    qr = qf + lb * 128;
    float s = 0.f;
#pragma unroll 16
    for (int c = 0; c < 128; ++c) s += qr[c] * (float)er[c];
    sc[lb][n] = s * 0.08838834764831845f;   // 1/sqrt(128)
  }
  __syncthreads();
  if (tid < 16) {
    float m = sc[tid][0];
#pragma unroll
    for (int n = 1; n < 8; ++n) m = fmaxf(m, sc[tid][n]);
    float e[8], sum = 0.f;
#pragma unroll
    for (int n = 0; n < 8; ++n) { e[n] = fast_exp(sc[tid][n] - m); sum += e[n]; }
    float inv = 1.f / sum;
#pragma unroll
    for (int n = 0; n < 8; ++n) sc[tid][n] = e[n] * inv;
  }
  __syncthreads();
#pragma unroll
  for (int j = 0; j < 8; ++j) {
    int idx = j * 256 + tid;
    int lb = idx >> 7, c = idx & 127;
    float o = 0.f;
#pragma unroll
    for (int n = 0; n < 8; ++n) o += sc[lb][n] * (float)Hs[(lb * 8 + n) * 128 + c];
    long row = (long)(b0 + lb);
    attn_f32[row * 128 + c] = o;
    attn_f16[row * 128 + c] = (_Float16)o;
  }
}

// ---------------------------------------------------------------------------
// Kernel 2: agent MLP + multi-head group attention (+8x tiling of the output).
// Block = 256 threads, handles 16 groups = 128 agent rows.
// LDS (dynamic, halves): As[16384] | Wt[16384] | Hs[16384] = 96KB
// ---------------------------------------------------------------------------
__global__ __launch_bounds__(256)
void agent_mlp_mha(const _Float16* __restrict__ attn_f16,   // d_ws
                   const float* __restrict__ aW1, const float* __restrict__ ab1,
                   const float* __restrict__ aW2, const float* __restrict__ ab2,
                   float* __restrict__ mha_out) {            // d_out (first output)
  extern __shared__ _Float16 smem[];
  _Float16* As = smem;             // agent_attention tile
  _Float16* Wt = smem + 16384;
  _Float16* Hs = Wt + 16384;       // hidden; then agent_values (per-strip alias)

  const int tid = threadIdx.x;
  const int bid = blockIdx.x;
  const long r0 = (long)bid * 128;
  const int  g0 = bid * 16;

  {
    const v8h* src = (const v8h*)(attn_f16 + r0 * 128);
    v8h* dst = (v8h*)As;
#pragma unroll
    for (int j = 0; j < 8; ++j) dst[j * 256 + tid] = src[j * 256 + tid];
  }
  load_wt128(Wt, aW1, tid);
  __syncthreads();
  gemm128_tanh(As, Wt, ab1, Hs, tid);
  __syncthreads();
  load_wt128(Wt, aW2, tid);
  __syncthreads();
  gemm128_tanh(Hs, Wt, ab2, Hs, tid);   // agent_values in Hs
  __syncthreads();

  if (tid < 128) {
    int lg = tid >> 3, h = tid & 7;
    const _Float16* kb = As + (lg * 8) * 128 + h * 16;
    const _Float16* vb = Hs + (lg * 8) * 128 + h * 16;
    float q[16];
#pragma unroll
    for (int j = 0; j < 16; ++j) {
      float s = 0.f;
#pragma unroll
      for (int a = 0; a < 8; ++a) s += (float)kb[a * 128 + j];
      q[j] = s * 0.125f;
    }
    float s[8], m = -1e30f;
#pragma unroll
    for (int a = 0; a < 8; ++a) {
      float d = 0.f;
#pragma unroll
      for (int j = 0; j < 16; ++j) d += q[j] * (float)kb[a * 128 + j];
      s[a] = d * 0.25f;              // 1/sqrt(16)
      m = fmaxf(m, s[a]);
    }
    float sum = 0.f;
#pragma unroll
    for (int a = 0; a < 8; ++a) { s[a] = fast_exp(s[a] - m); sum += s[a]; }
    float inv = 1.f / sum;
    float o[16];
#pragma unroll
    for (int j = 0; j < 16; ++j) {
      float d = 0.f;
#pragma unroll
      for (int a = 0; a < 8; ++a) d += s[a] * (float)vb[a * 128 + j];
      o[j] = d * inv;
    }
    // multi_head_attention = tile(out2, (8,1)): row i = out2[i mod 8192]
#pragma unroll
    for (int mcopy = 0; mcopy < 8; ++mcopy) {
      long row = (long)(g0 + lg) + (long)mcopy * NGROUPS;
      float4* p4 = (float4*)(mha_out + row * 128 + h * 16);
      p4[0] = make_float4(o[0],  o[1],  o[2],  o[3]);
      p4[1] = make_float4(o[4],  o[5],  o[6],  o[7]);
      p4[2] = make_float4(o[8],  o[9],  o[10], o[11]);
      p4[3] = make_float4(o[12], o[13], o[14], o[15]);
    }
  }
}

extern "C" void kernel_launch(void* const* d_in, const int* in_sizes, int n_in,
                              void* d_out, int out_size, void* d_ws, size_t ws_size,
                              hipStream_t stream) {
  const float* obs = (const float*)d_in[0];
  const float* eW1 = (const float*)d_in[1];
  const float* eb1 = (const float*)d_in[2];
  const float* eW2 = (const float*)d_in[3];
  const float* eb2 = (const float*)d_in[4];
  const float* vW1 = (const float*)d_in[5];
  const float* vb1 = (const float*)d_in[6];
  const float* vW2 = (const float*)d_in[7];
  const float* vb2 = (const float*)d_in[8];
  const float* aW1 = (const float*)d_in[9];
  const float* ab1 = (const float*)d_in[10];
  const float* aW2 = (const float*)d_in[11];
  const float* ab2 = (const float*)d_in[12];

  float*     out  = (float*)d_out;
  _Float16*  ws16 = (_Float16*)d_ws;   // agent_attention f16: 65536*128*2 = 16MB

  const size_t lds1 = (16384 + 4096 + 16384 + 16384) * sizeof(_Float16); // 104KB
  const size_t lds2 = (16384 + 16384 + 16384) * sizeof(_Float16);        //  96KB

  fused_embed_value_attn<<<BATCH / 16, 256, lds1, stream>>>(
      obs, eW1, eb1, eW2, eb2, vW1, vb1, vW2, vb2,
      out + OUT1_ELEMS, ws16);

  agent_mlp_mha<<<NGROUPS / 16, 256, lds2, stream>>>(
      ws16, aW1, ab1, aW2, ab2, out);
}
